// FusedOnlineAttention_8701603741984
// MI455X (gfx1250) — matrix operand order repaired and checked by reading, not compiled, not run on hardware
//
#include <hip/hip_runtime.h>

typedef __attribute__((ext_vector_type(16))) _Float16 v16h;
typedef __attribute__((ext_vector_type(8)))  float    v8f;

#define SEQ   2048
#define NH    16
#define DH    128
#define QTILE 64
#define KTILE 32
#define KPAD  8
#define LOG2E 1.4426950408889634f

// XOR-butterfly within a 16-lane DPP row: pure VALU, no DS-pipe traffic.
template<int XMASK>
__device__ __forceinline__ float row_xor_f32(float x) {
    int r = __builtin_amdgcn_update_dpp(0, __builtin_bit_cast(int, x),
                                        0x160 | XMASK /*ROW_XMASK*/, 0xf, 0xf, true);
    return __builtin_bit_cast(float, r);
}

// v_cvt_pk_rtz_f16_f32: two f32 -> packed dword of two f16.
__device__ __forceinline__ unsigned pkrtz(float x, float y) {
    auto t = __builtin_amdgcn_cvt_pkrtz(x, y);   // __fp16 ext_vector(2)
    return __builtin_bit_cast(unsigned, t);
}

// Load 16 packed halfs (one 16-bit WMMA operand for this lane) as two 16B chunks.
__device__ __forceinline__ v16h ld_b16x16(const _Float16* p0, const _Float16* p1) {
    v16h r;
    reinterpret_cast<uint4*>(&r)[0] = *reinterpret_cast<const uint4*>(p0);
    reinterpret_cast<uint4*>(&r)[1] = *reinterpret_cast<const uint4*>(p1);
    return r;
}

__global__ __launch_bounds__(128, 1)
void fa_alibi_fwd(const float* __restrict__ q, const float* __restrict__ k,
                  const float* __restrict__ v, const float* __restrict__ slopes,
                  float* __restrict__ out)
{
    __shared__ _Float16 Ql[QTILE][DH + KPAD];           // 64 x 136 f16
    __shared__ _Float16 Kl[2][KTILE][DH + KPAD];        // double-buffered [key][d]
    __shared__ _Float16 Vt[2][DH][KTILE + KPAD];        // double-buffered, transposed + key-permuted
    __shared__ _Float16 Pl[4][16][KTILE + KPAD];        // per-wave P scratch [m][permuted col]

    const int h    = blockIdx.y;
    const int q0   = blockIdx.x * QTILE;
    const int tid  = threadIdx.x;
    const int wave = tid >> 5;
    const int lane = tid & 31;
    const int g    = lane >> 4;    // half-wave group
    const int ln   = lane & 15;

    // Log2 domain: fold 1/sqrt(D)*log2(e) into Q, log2(e) into the slope.
    const float qk_scale = 0.08838834764831845f * LOG2E;
    const float slope2   = slopes[h] * LOG2E;

    // ---- stage Q tile: 64x128 f32 -> f16 (pre-scaled) in LDS ----
    for (int it = 0; it < 16; ++it) {
        int i   = tid + 128 * it;
        int row = i >> 5;            // 32 float4 per row
        int d   = (i & 31) << 2;
        const float4 qv = *reinterpret_cast<const float4*>(
            q + (((size_t)(q0 + row) * NH + h) * DH + d));
        uint2 u;
        u.x = pkrtz(qv.x * qk_scale, qv.y * qk_scale);
        u.y = pkrtz(qv.z * qk_scale, qv.w * qk_scale);
        *reinterpret_cast<uint2*>(&Ql[row][d]) = u;
    }
    __syncthreads();

    // ---- per-wave persistent Q A-fragments (16x32 per K-chunk of D) ----
    const int qrow = 16 * wave + ln;             // A layout: lane holds row ln
    v16h qfrag[4];
#pragma unroll
    for (int kc = 0; kc < 4; ++kc) {
        const int db = 32 * kc;
        qfrag[kc] = ld_b16x16(&Ql[qrow][db + 8 * g], &Ql[qrow][db + 16 + 8 * g]);
    }

    // ---- online-softmax state and output accumulators ----
    float mrow[8], lrow[8];
    v8f acc[8];
#pragma unroll
    for (int r = 0; r < 8; ++r) { mrow[r] = -1e30f; lrow[r] = 0.f; }
#pragma unroll
    for (int dt = 0; dt < 8; ++dt) acc[dt] = (v8f){};

    const int row_lo = q0 + 16 * wave + 8 * g;   // C layout: VGPR r -> q row row_lo+r
    const int jtiles = 2 * blockIdx.x + 2;       // causal bound, uniform across block

    // ---- register prefetch (double-buffer the K/V LDS tiles) ----
    float4 kpre[8], vpre[8];
    auto prefetch = [&](int j) {
        const int kb = j * KTILE;
#pragma unroll
        for (int it = 0; it < 8; ++it) {
            int i   = tid + 128 * it;
            int key = i >> 5;
            int d   = (i & 31) << 2;
            size_t gofs = (((size_t)(kb + key) * NH + h) * DH + d);
            kpre[it] = *reinterpret_cast<const float4*>(k + gofs);
            vpre[it] = *reinterpret_cast<const float4*>(v + gofs);
        }
    };
    auto store_tile = [&](int buf) {
#pragma unroll
        for (int it = 0; it < 8; ++it) {
            int i   = tid + 128 * it;
            int key = i >> 5;
            int d   = (i & 31) << 2;
            uint2 u;
            u.x = pkrtz(kpre[it].x, kpre[it].y);
            u.y = pkrtz(kpre[it].z, kpre[it].w);
            *reinterpret_cast<uint2*>(&Kl[buf][key][d]) = u;
            // V transposed and key-permuted (key ln -> col 2*ln, key 16+ln -> col 2*ln+1)
            union { unsigned u; _Float16 hh[2]; } ua, ub;
            ua.u = pkrtz(vpre[it].x, vpre[it].y);
            ub.u = pkrtz(vpre[it].z, vpre[it].w);
            int scol = ((key & 15) << 1) | (key >> 4);
            Vt[buf][d + 0][scol] = ua.hh[0];
            Vt[buf][d + 1][scol] = ua.hh[1];
            Vt[buf][d + 2][scol] = ub.hh[0];
            Vt[buf][d + 3][scol] = ub.hh[1];
        }
    };

    prefetch(0);
    for (int j = 0; j < jtiles; ++j) {
        const int kbase = j * KTILE;
        const int buf   = j & 1;

        store_tile(buf);             // regs (tile j) -> LDS buffer j&1
        __syncthreads();

        // ---- scores S = Q K^T : batch all 8 K B-fragments, then 8 WMMAs ----
        v16h bf[8];
#pragma unroll
        for (int s = 0; s < 8; ++s) {
            const int nt = s >> 2, kc = s & 3;
            const _Float16* p = &Kl[buf][16 * nt + ln][32 * kc + 8 * g];
            bf[s] = ld_b16x16(p, p + 16);
        }
        v8f sc[2] = { (v8f){}, (v8f){} };
#pragma unroll
        for (int s = 0; s < 8; ++s)
            sc[s >> 2] = __builtin_amdgcn_wmma_f32_16x16x32_f16(
                             false, qfrag[s & 3], false, bf[s], (short)0, sc[s >> 2],
                             false, false);

        if (j + 1 < jtiles) prefetch(j + 1);   // HBM/L2 latency overlaps softmax+PV

        // ---- ALiBi bias (+ causal mask only on boundary tiles) ----
        const int dcol0 = kbase + ln - row_lo;            // col-row at nt=0, r=0
        const bool interior = (kbase + KTILE) <= (q0 + 16 * wave);  // wave-uniform
        if (interior) {
#pragma unroll
            for (int nt = 0; nt < 2; ++nt) {
                float bs = slope2 * (float)(dcol0 + 16 * nt);
#pragma unroll
                for (int r = 0; r < 8; ++r) {
                    sc[nt][r] += bs;
                    bs -= slope2;
                }
            }
        } else {
#pragma unroll
            for (int nt = 0; nt < 2; ++nt) {
                const int d0 = dcol0 + 16 * nt;
                float bs = slope2 * (float)d0;
#pragma unroll
                for (int r = 0; r < 8; ++r) {
                    float s = sc[nt][r] + bs;
                    bs -= slope2;
                    sc[nt][r] = (d0 > r) ? -1e30f : s;
                }
            }
        }

        // ---- online softmax (log2 domain), DPP row_xmask reductions ----
        float pr0[8], pr1[8], corr[8];
#pragma unroll
        for (int r = 0; r < 8; ++r) {
            float mx = fmaxf(sc[0][r], sc[1][r]);
            mx = fmaxf(mx, row_xor_f32<1>(mx));
            mx = fmaxf(mx, row_xor_f32<2>(mx));
            mx = fmaxf(mx, row_xor_f32<4>(mx));
            mx = fmaxf(mx, row_xor_f32<8>(mx));
            const float mn = fmaxf(mrow[r], mx);
            corr[r] = __builtin_amdgcn_exp2f(mrow[r] - mn);
            mrow[r] = mn;
            pr0[r] = __builtin_amdgcn_exp2f(sc[0][r] - mn);
            pr1[r] = __builtin_amdgcn_exp2f(sc[1][r] - mn);
            float rs = pr0[r] + pr1[r];
            rs += row_xor_f32<1>(rs);
            rs += row_xor_f32<2>(rs);
            rs += row_xor_f32<4>(rs);
            rs += row_xor_f32<8>(rs);
            lrow[r] = lrow[r] * corr[r] + rs;
        }
#pragma unroll
        for (int dt = 0; dt < 8; ++dt)
#pragma unroll
            for (int r = 0; r < 8; ++r) acc[dt][r] *= corr[r];

        // ---- P -> LDS (packed dword per lane, key-permuted columns) ----
#pragma unroll
        for (int r = 0; r < 8; ++r) {
            *reinterpret_cast<unsigned*>(&Pl[wave][r + 8 * g][2 * ln]) =
                pkrtz(pr0[r], pr1[r]);
        }
        asm volatile("" ::: "memory");   // LDS is in-order per wave; pin store->load order
        v16h pa = ld_b16x16(&Pl[wave][ln][8 * g], &Pl[wave][ln][16 + 8 * g]);

        // ---- O += P V : batch all 8 V B-fragments, then 8 WMMAs ----
        v16h vf[8];
#pragma unroll
        for (int dt = 0; dt < 8; ++dt) {
            const _Float16* p = &Vt[buf][16 * dt + ln][8 * g];
            vf[dt] = ld_b16x16(p, p + 16);
        }
#pragma unroll
        for (int dt = 0; dt < 8; ++dt)
            acc[dt] = __builtin_amdgcn_wmma_f32_16x16x32_f16(
                          false, pa, false, vf[dt], (short)0, acc[dt], false, false);
    }

    // ---- epilogue: normalize, store fp32 ----
#pragma unroll
    for (int r = 0; r < 8; ++r) {
        const float inv = 1.0f / lrow[r];
        const size_t row = (size_t)(row_lo + r);
#pragma unroll
        for (int dt = 0; dt < 8; ++dt) {
            out[(row * NH + h) * DH + 16 * dt + ln] = acc[dt][r] * inv;
        }
    }
}

extern "C" void kernel_launch(void* const* d_in, const int* in_sizes, int n_in,
                              void* d_out, int out_size, void* d_ws, size_t ws_size,
                              hipStream_t stream) {
    (void)in_sizes; (void)n_in; (void)out_size; (void)d_ws; (void)ws_size;
    const float* q  = (const float*)d_in[0];
    const float* k  = (const float*)d_in[1];
    const float* v  = (const float*)d_in[2];
    const float* sl = (const float*)d_in[3];
    float* o = (float*)d_out;
    dim3 grid(SEQ / QTILE, NH);
    fa_alibi_fwd<<<grid, 128, 0, stream>>>(q, k, v, sl, o);
}
